// WindowCrossAttention_34591666602373
// MI455X (gfx1250) — compile-verified, matrix operand order checked by reading, not run on hardware
//
#include <hip/hip_runtime.h>
#include <cstdint>

typedef __attribute__((ext_vector_type(16))) __bf16 v16bf;
typedef __attribute__((ext_vector_type(8)))  __bf16 v8bf;
typedef __attribute__((ext_vector_type(8)))  float  v8f;

#define DIM    256
#define NTOK   64
#define HEADS  8
#define DH     32
#define NW     1024
#define KBLK   8            // 256 / 32 k-steps for the projections
#define XSTR   264          // bf16 row stride for X/Q/K (528 B, 16B aligned)
#define VTSTR  72           // bf16 row stride for V^T (144 B, 16B aligned)
#define SSTR   64           // f32 score row stride (scalar access only)
#define PSTR   72           // bf16 prob row stride (144 B, 16B aligned)
#define SCALE  0.17677669529663689f   // 32^-0.5

__device__ __forceinline__ v8f wmma_bf16(v16bf a, v16bf b, v8f c) {
    return __builtin_amdgcn_wmma_f32_16x16x32_bf16(false, a, false, b, (short)0, c,
                                                   false, false);
}

// 16-bit A-fragment (or B-fragment of an operand that is stored transposed):
// lane l picks row row0 + (l&15); two contiguous 16B chunks at
// k0 + (l>>4)*8 and +16 further  (documented 16-bit A layout).
__device__ __forceinline__ v16bf lds_frag_row(const __bf16* base, int stride,
                                              int row0, int k0, int lane) {
    const int half = lane >> 4;
    const int row  = row0 + (lane & 15);
    const __bf16* p = base + (size_t)row * stride + k0 + half * 8;
    v8bf c0 = *(const v8bf*)(p);
    v8bf c1 = *(const v8bf*)(p + 16);
    return __builtin_shufflevector(c0, c1, 0,1,2,3,4,5,6,7,8,9,10,11,12,13,14,15);
}

// B-fragment from pre-packed weights: 512 bf16 per (nb,kb) tile, lane-major.
__device__ __forceinline__ v16bf ldg_fragB(const __bf16* wpk, int nb, int kb, int lane) {
    const __bf16* p = wpk + ((((size_t)nb * KBLK) + kb) << 9) + lane * 16;
    v8bf c0 = *(const v8bf*)(p);
    v8bf c1 = *(const v8bf*)(p + 8);
    return __builtin_shufflevector(c0, c1, 0,1,2,3,4,5,6,7,8,9,10,11,12,13,14,15);
}

// GEMM: 64xK (LDS bf16, stride XSTR) x packed W -> acc[NTJ*4] tiles per wave.
template<int NTJ>
__device__ __forceinline__ void proj_accum(const __bf16* sA, const __bf16* wpk,
                                           int wave, int lane, v8f* acc) {
    const v8f vz = {};
    #pragma unroll
    for (int i = 0; i < NTJ * 4; ++i) acc[i] = vz;
    for (int kb = 0; kb < KBLK; ++kb) {
        v16bf a[4];
        #pragma unroll
        for (int ti = 0; ti < 4; ++ti)
            a[ti] = lds_frag_row(sA, XSTR, ti * 16, kb * 32, lane);
        #pragma unroll
        for (int j = 0; j < NTJ; ++j) {
            v16bf bb = ldg_fragB(wpk, wave * NTJ + j, kb, lane);
            #pragma unroll
            for (int ti = 0; ti < 4; ++ti)
                acc[j * 4 + ti] = wmma_bf16(a[ti], bb, acc[j * 4 + ti]);
        }
    }
}

// Pack weight (K x Nout, f32 row-major) into bf16 fragment-major layout.
__global__ void pack_w(const float* __restrict__ src, __bf16* __restrict__ dst, int Nout) {
    int g    = blockIdx.x * 256 + threadIdx.x;
    int tile = g >> 9;
    int r    = g & 511;
    int lane = r >> 4;
    int e    = r & 15;
    int nb   = tile / KBLK;
    int kb   = tile % KBLK;
    int k    = kb * 32 + (lane >> 4) * 8 + (e & 7) + ((e >> 3) << 4);
    int n    = nb * 16 + (lane & 15);
    dst[g] = (__bf16)src[(size_t)k * Nout + n];
}

// Pre-gather relative-position bias: bias_full[h][m][n] = table[rel_idx[m,n]][h].
__global__ void pack_bias(const float* __restrict__ table, const int* __restrict__ rel_idx,
                          float* __restrict__ dst) {
    int g = blockIdx.x * 256 + threadIdx.x;        // 8*64*64 = 32768 total
    int h = g >> 12;
    int mn = g & 4095;
    dst[g] = table[rel_idx[mn] * HEADS + h];
}

__global__ __launch_bounds__(256, 1)
void win_xattn_fused(const float* __restrict__ x0, const float* __restrict__ x1,
                     const float* __restrict__ mask, const float* __restrict__ biasf,
                     const float* __restrict__ bq, const float* __restrict__ bkv,
                     const float* __restrict__ bp,
                     const __bf16* __restrict__ wqpk, const __bf16* __restrict__ wkvpk,
                     const __bf16* __restrict__ wppk, float* __restrict__ out) {
    __shared__ __bf16 sX [NTOK * XSTR];   // x staging, later O (bf16)
    __shared__ __bf16 sQ [NTOK * XSTR];
    __shared__ __bf16 sK [NTOK * XSTR];
    __shared__ __bf16 sVT[DIM  * VTSTR];  // V transposed: [channel][token]
    __shared__ float  sS [NTOK * SSTR];
    __shared__ __bf16 sP [NTOK * PSTR];

    const int tid  = threadIdx.x;
    const int lane = tid & 31;
    const int wave = tid >> 5;
    const int half = lane >> 4;
    const int ln   = lane & 15;
    const int b    = blockIdx.x;

    // ---- stage x1 (f32 -> bf16 LDS) ----
    {
        const float* src = x1 + (size_t)b * NTOK * DIM;
        for (int i = tid; i < NTOK * DIM; i += 256)
            sX[(i >> 8) * XSTR + (i & 255)] = (__bf16)src[i];
    }
    __syncthreads();

    // ---- Q = (x1 @ Wq + bq) * scale -> sQ ----
    {
        v8f acc[8];
        proj_accum<2>(sX, wqpk, wave, lane, acc);
        #pragma unroll
        for (int j = 0; j < 2; ++j) {
            const int col = (wave * 2 + j) * 16 + ln;
            const float bv = bq[col];
            #pragma unroll
            for (int ti = 0; ti < 4; ++ti)
                #pragma unroll
                for (int r = 0; r < 8; ++r) {
                    int tok = ti * 16 + half * 8 + r;
                    sQ[tok * XSTR + col] = (__bf16)((acc[j * 4 + ti][r] + bv) * SCALE);
                }
        }
    }
    __syncthreads();

    // ---- stage x0 ----
    {
        const float* src = x0 + (size_t)b * NTOK * DIM;
        for (int i = tid; i < NTOK * DIM; i += 256)
            sX[(i >> 8) * XSTR + (i & 255)] = (__bf16)src[i];
    }
    __syncthreads();

    // ---- K,V = x0 @ Wkv + bkv : K row-major -> sK, V transposed -> sVT ----
    {
        v8f acc[16];
        proj_accum<4>(sX, wkvpk, wave, lane, acc);
        #pragma unroll
        for (int j = 0; j < 4; ++j) {
            const int col = (wave * 4 + j) * 16 + ln;
            const float bv = bkv[col];
            #pragma unroll
            for (int ti = 0; ti < 4; ++ti) {
                if (col < DIM) {
                    #pragma unroll
                    for (int r = 0; r < 8; ++r) {
                        int tok = ti * 16 + half * 8 + r;
                        sK[tok * XSTR + col] = (__bf16)(acc[j * 4 + ti][r] + bv);
                    }
                } else {
                    // contiguous 8 tokens per lane -> single ds_store_b128
                    v8bf vv;
                    #pragma unroll
                    for (int r = 0; r < 8; ++r)
                        vv[r] = (__bf16)(acc[j * 4 + ti][r] + bv);
                    *(v8bf*)(&sVT[(size_t)(col - DIM) * VTSTR + ti * 16 + half * 8]) = vv;
                }
            }
        }
    }
    __syncthreads();

    // ---- attention heads ----
    const float* maskw = mask + (size_t)(b & (NW - 1)) * NTOK * NTOK;
    for (int h = 0; h < HEADS; ++h) {
        const float* bh = biasf + (size_t)h * NTOK * NTOK;
        // S = Qh @ Kh^T + bias + mask  (16 tiles, 2 per wave)
        {
            const int ti = wave & 3;
            #pragma unroll
            for (int jj = 0; jj < 2; ++jj) {
                const int tj = (wave >> 2) * 2 + jj;
                v16bf a  = lds_frag_row(sQ, XSTR, ti * 16, h * DH, lane);
                v16bf bb = lds_frag_row(sK, XSTR, tj * 16, h * DH, lane);
                v8f acc = {};
                acc = wmma_bf16(a, bb, acc);
                #pragma unroll
                for (int r = 0; r < 8; ++r) {
                    int m = ti * 16 + half * 8 + r;
                    int n = tj * 16 + ln;
                    sS[m * SSTR + n] = acc[r] + bh[m * NTOK + n] + maskw[m * NTOK + n];
                }
            }
        }
        __syncthreads();

        // row softmax: 4 threads per row, 16 cols each, shuffle-reduce (wave32)
        {
            const int row = tid >> 2;
            const int c0  = (tid & 3) * 16;
            float* srow = &sS[row * SSTR];
            float mx = -3.4e38f;
            #pragma unroll
            for (int j = 0; j < 16; ++j) mx = fmaxf(mx, srow[c0 + j]);
            mx = fmaxf(mx, __shfl_xor(mx, 1, 32));
            mx = fmaxf(mx, __shfl_xor(mx, 2, 32));
            float sum = 0.f;
            float e[16];
            #pragma unroll
            for (int j = 0; j < 16; ++j) { e[j] = __expf(srow[c0 + j] - mx); sum += e[j]; }
            sum += __shfl_xor(sum, 1, 32);
            sum += __shfl_xor(sum, 2, 32);
            const float inv = 1.f / sum;
            v8bf p0, p1;
            #pragma unroll
            for (int j = 0; j < 8; ++j) { p0[j] = (__bf16)(e[j] * inv); p1[j] = (__bf16)(e[8 + j] * inv); }
            *(v8bf*)(&sP[row * PSTR + c0])     = p0;
            *(v8bf*)(&sP[row * PSTR + c0 + 8]) = p1;
        }
        __syncthreads();

        // O_h = P @ V_h  (8 tiles, 1 per wave; K = 64 -> 2 WMMA steps)
        {
            const int ti = wave & 3;
            const int tj = wave >> 2;
            v8f acc = {};
            #pragma unroll
            for (int ks = 0; ks < 2; ++ks) {
                v16bf a  = lds_frag_row(sP,  PSTR,  ti * 16,            ks * 32, lane);
                v16bf bb = lds_frag_row(sVT, VTSTR, h * DH + tj * 16,   ks * 32, lane);
                acc = wmma_bf16(a, bb, acc);
            }
            #pragma unroll
            for (int r = 0; r < 8; ++r) {
                int m = ti * 16 + half * 8 + r;
                sX[m * XSTR + h * DH + tj * 16 + ln] = (__bf16)acc[r];
            }
        }
        __syncthreads();
    }

    // ---- out projection: Y = O @ Wp + bp -> global f32 ----
    {
        v8f acc[8];
        proj_accum<2>(sX, wppk, wave, lane, acc);
        float* gout = out + (size_t)b * NTOK * DIM;
        #pragma unroll
        for (int j = 0; j < 2; ++j) {
            const int col = (wave * 2 + j) * 16 + ln;
            const float bv = bp[col];
            #pragma unroll
            for (int ti = 0; ti < 4; ++ti)
                #pragma unroll
                for (int r = 0; r < 8; ++r) {
                    int tok = ti * 16 + half * 8 + r;
                    gout[(size_t)tok * DIM + col] = acc[j * 4 + ti][r] + bv;
                }
        }
    }
}

extern "C" void kernel_launch(void* const* d_in, const int* in_sizes, int n_in,
                              void* d_out, int out_size, void* d_ws, size_t ws_size,
                              hipStream_t stream) {
    const float* x0         = (const float*)d_in[0];
    const float* x1         = (const float*)d_in[1];
    const float* mask       = (const float*)d_in[2];
    const int*   rel_idx    = (const int*)d_in[3];
    const float* Wq         = (const float*)d_in[4];
    const float* bq         = (const float*)d_in[5];
    const float* Wkv        = (const float*)d_in[6];
    const float* bkv        = (const float*)d_in[7];
    const float* bias_table = (const float*)d_in[8];
    const float* Wp         = (const float*)d_in[9];
    const float* bp         = (const float*)d_in[10];
    float* out = (float*)d_out;

    __bf16* wqpk  = (__bf16*)d_ws;                 // 256*256 bf16
    __bf16* wkvpk = wqpk + 256 * 256;              // 256*512 bf16
    __bf16* wppk  = wkvpk + 256 * 512;             // 256*256 bf16
    float*  biasf = (float*)(wppk + 256 * 256);    // 8*64*64 f32

    pack_w<<<256, 256, 0, stream>>>(Wq,  wqpk,  256);
    pack_w<<<512, 256, 0, stream>>>(Wkv, wkvpk, 512);
    pack_w<<<256, 256, 0, stream>>>(Wp,  wppk,  256);
    pack_bias<<<128, 256, 0, stream>>>(bias_table, rel_idx, biasf);

    win_xattn_fused<<<2048, 256, 0, stream>>>(x0, x1, mask, biasf,
                                              bq, bkv, bp,
                                              wqpk, wkvpk, wppk, out);
}
